// HierarchicalPCN_28149215657990
// MI455X (gfx1250) — matrix operand order repaired and checked by reading, not compiled, not run on hardware
//
#include <hip/hip_runtime.h>
#include <stdint.h>

// ---------------------------------------------------------------------------
// HierarchicalPCN inference on MI455X (gfx1250), bf16 WMMA + double-buffered
// async LDS staging. B=1024, L=1024, O=4096, 20 inference iterations.
// ---------------------------------------------------------------------------

typedef __bf16 bf16;
typedef bf16  v16bf __attribute__((ext_vector_type(16)));
typedef bf16  v8bf  __attribute__((ext_vector_type(8)));
typedef float v8f   __attribute__((ext_vector_type(8)));
typedef float v4f   __attribute__((ext_vector_type(4)));
typedef int   v4i_vs __attribute__((vector_size(16)));  // builtin's b128 unit

#define B_DIM 1024
#define L_DIM 1024
#define O_DIM 4096
#define ITERS 20
#define INF_LR 0.01f
#define SPARSE_Z 1e-3f

#define BM 64
#define BN 256
#define BK 32
#define THREADS 256
#define ASZ (BM * BK)   // A tile elements per buffer
#define BSZ (BN * BK)   // B tile elements per buffer

#if __has_builtin(__builtin_amdgcn_global_load_async_to_lds_b128)
#define HAVE_ASYNC 1
#endif

// 16-byte global->LDS copy; async (ASYNCcnt-tracked) when available.
__device__ __forceinline__ void copy16_g2l(bf16* lds, const bf16* g) {
#ifdef HAVE_ASYNC
  __builtin_amdgcn_global_load_async_to_lds_b128(
      (__attribute__((address_space(1))) v4i_vs*)(uintptr_t)g,
      (__attribute__((address_space(3))) v4i_vs*)(uint32_t)(uintptr_t)lds,
      0, 0);
#else
  *(v8bf*)lds = *(const v8bf*)g;
#endif
}

template <int N>
__device__ __forceinline__ void async_wait_le() {
#ifdef HAVE_ASYNC
#if __has_builtin(__builtin_amdgcn_s_wait_asynccnt)
  __builtin_amdgcn_s_wait_asynccnt(N);
#else
  asm volatile("s_wait_asynccnt %0" ::"i"(N) : "memory");
#endif
#endif
}

// --- WMMA fragment loads from LDS (tiles stored with row stride BK bf16) ---
// A (16x32 bf16): lane m in 0..15 holds K{0..7,16..23}; lane m+16: K{8..15,24..31}
__device__ __forceinline__ v16bf load_frag_a(const bf16* s, int row16, int lane) {
  int r  = row16 + (lane & 15);
  int kb = (lane < 16) ? 0 : 8;
  v8bf lo = *(const v8bf*)(s + r * BK + kb);
  v8bf hi = *(const v8bf*)(s + r * BK + kb + 16);
  v16bf out;
#pragma unroll
  for (int i = 0; i < 8; ++i) { out[i] = lo[i]; out[i + 8] = hi[i]; }
  return out;
}

// B (32x16 bf16), staged transposed in LDS as Bt[n][k]:
// lane n in 0..15 holds K{0..7,16..23} of column n; lane n+16 holds K{8..15,24..31}.
__device__ __forceinline__ v16bf load_frag_b(const bf16* s, int col16, int lane) {
  int r  = col16 + (lane & 15);
  int kb = (lane < 16) ? 0 : 8;
  v8bf lo = *(const v8bf*)(s + r * BK + kb);
  v8bf hi = *(const v8bf*)(s + r * BK + kb + 16);
  v16bf out;
#pragma unroll
  for (int i = 0; i < 8; ++i) { out[i] = lo[i]; out[i + 8] = hi[i]; }
  return out;
}

__device__ __forceinline__ v8f wmma_bf16(v16bf a, v16bf b, v8f c) {
  return __builtin_amdgcn_wmma_f32_16x16x32_bf16(false, a, false, b,
                                                 (short)0, c, false, false);
}

// ---------------------------------------------------------------------------
// Prep kernels
// ---------------------------------------------------------------------------
__global__ void prep_w(const float* __restrict__ W, bf16* __restrict__ Wbf,
                       bf16* __restrict__ Wt) {
  size_t idx = (size_t)blockIdx.x * 256 + threadIdx.x;  // over O*L
  size_t o = idx / L_DIM, l = idx % L_DIM;
  bf16 v = (bf16)W[idx];
  Wbf[idx] = v;
  Wt[l * O_DIM + o] = v;
}

__global__ void init_z(const float* __restrict__ mu, float* __restrict__ z) {
  size_t idx = (size_t)blockIdx.x * 256 + threadIdx.x;  // over B*L
  z[idx] = mu[idx % L_DIM];
}

// ---------------------------------------------------------------------------
// Forward: G[b,o] = (1 - p^2) * (x[b,o] - p),  p = tanh(sum_l z[b,l]*W[o,l])
// GEMM M=B, N=O, K=L.  A = z (fp32 -> bf16 on stage), B-source = Wbf rows n.
// Wave tile 32x64 (2x4 frags, 8 WMMA / K-step); block tile 64x256.
// ---------------------------------------------------------------------------
__global__ __launch_bounds__(THREADS) void pcn_forward(
    const float* __restrict__ z, const bf16* __restrict__ Wbf,
    const float* __restrict__ x, bf16* __restrict__ G) {
  __shared__ __align__(16) bf16 sA[2 * ASZ];
  __shared__ __align__(16) bf16 sB[2 * BSZ];

  const int tid = threadIdx.x, lane = tid & 31, wave = tid >> 5;
  const int waveM = wave >> 2, waveN = wave & 3;
  const int m0 = blockIdx.y * BM, n0 = blockIdx.x * BN;
  const int wm = waveM * 32, wn = waveN * 64;

  v8f acc[2][4] = {};

  const int arow = tid >> 2, acb = (tid & 3) * 8;  // A chunk: 8 elems

  // Stage tiles for K-step `kt` into buffer `bi`. 4 async issues per thread.
  auto stage = [&](int bi, int kt) {
    // A: 64x32 fp32 -> bf16 through VGPRs (DScnt-tracked store)
    const float* zsrc = z + (size_t)(m0 + arow) * L_DIM + kt + acb;
    v4f f0 = *(const v4f*)zsrc;
    v4f f1 = *(const v4f*)(zsrc + 4);
    v8bf a;
#pragma unroll
    for (int i = 0; i < 4; ++i) { a[i] = (bf16)f0[i]; a[i + 4] = (bf16)f1[i]; }
    *(v8bf*)(sA + bi * ASZ + arow * BK + acb) = a;
    // B: 256 rows of Wbf, 32 bf16 each = 1024 chunks; 4 per thread, async
#pragma unroll
    for (int k = 0; k < 4; ++k) {
      int c = tid + k * THREADS;
      int brow = c >> 2, bcb = (c & 3) * 8;
      copy16_g2l(sB + bi * BSZ + brow * BK + bcb,
                 Wbf + (size_t)(n0 + brow) * L_DIM + kt + bcb);
    }
  };

  stage(0, 0);  // prologue prefetch
  const int NK = L_DIM / BK;
  for (int i = 0; i < NK; ++i) {
    const int bi = i & 1;
    __syncthreads();  // all waves done computing buffer bi^1 (previous step)
    if (i + 1 < NK) {
      stage(bi ^ 1, (i + 1) * BK);
      async_wait_le<4>();  // in-order: drains this buffer's 4 ops from step i-1
    } else {
      async_wait_le<0>();
    }
    __syncthreads();  // buffer bi fully populated across all waves

    const bf16* cA = sA + bi * ASZ;
    const bf16* cB = sB + bi * BSZ;
    v16bf a0 = load_frag_a(cA, wm + 0, lane);
    v16bf a1 = load_frag_a(cA, wm + 16, lane);
#pragma unroll
    for (int j = 0; j < 4; ++j) {
      v16bf b = load_frag_b(cB, wn + j * 16, lane);
      acc[0][j] = wmma_bf16(a0, b, acc[0][j]);
      acc[1][j] = wmma_bf16(a1, b, acc[1][j]);
    }
  }

  // Epilogue: tanh + error weighting -> G (bf16)
  const int lane15 = lane & 15;
  const int rofs = (lane < 16) ? 0 : 8;
#pragma unroll
  for (int i = 0; i < 2; ++i)
#pragma unroll
    for (int j = 0; j < 4; ++j)
#pragma unroll
      for (int r = 0; r < 8; ++r) {
        int m = m0 + wm + i * 16 + r + rofs;
        int n = n0 + wn + j * 16 + lane15;
        float p  = tanhf(acc[i][j][r]);
        float ex = x[(size_t)m * O_DIM + n] - p;
        G[(size_t)m * O_DIM + n] = (bf16)((1.0f - p * p) * ex);
      }
}

// ---------------------------------------------------------------------------
// Backward + update: z[b,l] -= lr*((z-mu) - sum_o G[b,o]*W[o,l] + sp*sign(z))
// GEMM M=B, N=L, K=O.  A = G (bf16, async), B-source = Wt rows n (Wt=W^T [L,O]).
// ---------------------------------------------------------------------------
__global__ __launch_bounds__(THREADS) void pcn_backward(
    const bf16* __restrict__ G, const bf16* __restrict__ Wt,
    const float* __restrict__ mu, float* __restrict__ z) {
  __shared__ __align__(16) bf16 sA[2 * ASZ];
  __shared__ __align__(16) bf16 sB[2 * BSZ];

  const int tid = threadIdx.x, lane = tid & 31, wave = tid >> 5;
  const int waveM = wave >> 2, waveN = wave & 3;
  const int m0 = blockIdx.y * BM, n0 = blockIdx.x * BN;
  const int wm = waveM * 32, wn = waveN * 64;

  v8f acc[2][4] = {};

  const int arow = tid >> 2, acb = (tid & 3) * 8;

  // 5 async issues per thread per stage (1 A + 4 B).
  auto stage = [&](int bi, int kt) {
    copy16_g2l(sA + bi * ASZ + arow * BK + acb,
               G + (size_t)(m0 + arow) * O_DIM + kt + acb);
#pragma unroll
    for (int k = 0; k < 4; ++k) {
      int c = tid + k * THREADS;
      int brow = c >> 2, bcb = (c & 3) * 8;
      copy16_g2l(sB + bi * BSZ + brow * BK + bcb,
                 Wt + (size_t)(n0 + brow) * O_DIM + kt + bcb);
    }
  };

  stage(0, 0);
  const int NK = O_DIM / BK;
  for (int i = 0; i < NK; ++i) {
    const int bi = i & 1;
    __syncthreads();
    if (i + 1 < NK) {
      stage(bi ^ 1, (i + 1) * BK);
      async_wait_le<5>();
    } else {
      async_wait_le<0>();
    }
    __syncthreads();

    const bf16* cA = sA + bi * ASZ;
    const bf16* cB = sB + bi * BSZ;
    v16bf a0 = load_frag_a(cA, wm + 0, lane);
    v16bf a1 = load_frag_a(cA, wm + 16, lane);
#pragma unroll
    for (int j = 0; j < 4; ++j) {
      v16bf b = load_frag_b(cB, wn + j * 16, lane);
      acc[0][j] = wmma_bf16(a0, b, acc[0][j]);
      acc[1][j] = wmma_bf16(a1, b, acc[1][j]);
    }
  }

  // Epilogue: z update (in place; each element owned by exactly one tile)
  const int lane15 = lane & 15;
  const int rofs = (lane < 16) ? 0 : 8;
#pragma unroll
  for (int i = 0; i < 2; ++i)
#pragma unroll
    for (int j = 0; j < 4; ++j)
#pragma unroll
      for (int r = 0; r < 8; ++r) {
        int m = m0 + wm + i * 16 + r + rofs;
        int n = n0 + wn + j * 16 + lane15;
        size_t idx = (size_t)m * L_DIM + n;
        float zv = z[idx];
        float sg = (zv > 0.0f) ? 1.0f : ((zv < 0.0f) ? -1.0f : 0.0f);
        float delta = (zv - mu[n]) - acc[i][j][r] + SPARSE_Z * sg;
        z[idx] = zv - INF_LR * delta;
      }
}

// ---------------------------------------------------------------------------
// Host launcher.  Inputs: x [B*O] f32, W [O*L] f32, mu [L] f32, inf_iters (=20,
// fixed by setup; hardcoded since a device scalar can't be read during graph
// capture).  d_out = z [B*L] f32.  Workspace: Wbf(8MB) | Wt(8MB) | G(8MB).
// ---------------------------------------------------------------------------
extern "C" void kernel_launch(void* const* d_in, const int* in_sizes, int n_in,
                              void* d_out, int out_size, void* d_ws, size_t ws_size,
                              hipStream_t stream) {
  const float* x  = (const float*)d_in[0];
  const float* W  = (const float*)d_in[1];
  const float* mu = (const float*)d_in[2];
  float* z = (float*)d_out;

  char* ws = (char*)d_ws;
  bf16* Wbf = (bf16*)ws;
  bf16* Wt  = (bf16*)(ws + (size_t)O_DIM * L_DIM * sizeof(bf16));
  bf16* G   = (bf16*)(ws + (size_t)O_DIM * L_DIM * sizeof(bf16) * 2);

  prep_w<<<(O_DIM * (size_t)L_DIM) / 256, 256, 0, stream>>>(W, Wbf, Wt);
  init_z<<<(B_DIM * (size_t)L_DIM) / 256, 256, 0, stream>>>(mu, z);

  dim3 gridF(O_DIM / BN, B_DIM / BM);  // 16 x 16
  dim3 gridB(L_DIM / BN, B_DIM / BM);  //  4 x 16
  for (int it = 0; it < ITERS; ++it) {
    pcn_forward<<<gridF, THREADS, 0, stream>>>(z, Wbf, x, G);
    pcn_backward<<<gridB, THREADS, 0, stream>>>(G, Wt, mu, z);
  }
}
